// FlowLevel_60567628808340
// MI455X (gfx1250) — matrix operand order repaired and checked by reading, not compiled, not run on hardware
//
#include <hip/hip_runtime.h>
#include <hip/hip_bf16.h>

#define DEPTH 10
#define WIDTH 128
#define NROWS (1u << 20)
#define WAVES_PER_BLOCK 8
#define TILES_PER_WAVE 2
#define TILES_PER_BLOCK (WAVES_PER_BLOCK * TILES_PER_WAVE)
#define ABUF_STRIDE 40   // halfs per row: 32 data + 8 pad, keeps 16B alignment

typedef __attribute__((ext_vector_type(16))) _Float16 v16h;
typedef __attribute__((ext_vector_type(8)))  float    v8f;

union FragAB {
    v16h     v;
    _Float16 h[16];
    int4     q[2];
};

// D = A(16x32 f16, samples x features) * B(32x16 f16, features x outfeat) + C(f32)
// A lane layout (ISA 7.12.2): lane<16 -> M=lane, K chunks [32kf..+7],[32kf+16..+23];
//                             lane>=16 -> M=lane-16, K chunks [32kf+8..+15],[32kf+24..+31]
// B lane layout: lane<16 -> N=lane, lane>=16 -> N=lane-16; same K chunks as A (per half-wave)
// C/D layout: lane&15 = N, reg r -> M = r + 8*(lane>=16)

__global__ __launch_bounds__(256) void flow_wmma_kernel(
    const float* __restrict__ x,
    const float* __restrict__ an_scale,   // [DEPTH][2]
    const float* __restrict__ an_bias,    // [DEPTH][2]
    const float* __restrict__ conv_w,     // [DEPTH][2][2]
    const float* __restrict__ fc1_w,      // [DEPTH][128][1]
    const float* __restrict__ fc1_b,      // [DEPTH][128]
    const float* __restrict__ an1_scale,  // [DEPTH][128]
    const float* __restrict__ an1_bias,   // [DEPTH][128]
    const float* __restrict__ fc2_w,      // [DEPTH][128][128]
    const float* __restrict__ fc2_b,      // [DEPTH][128]
    const float* __restrict__ an2_scale,  // [DEPTH][128]
    const float* __restrict__ an2_bias,   // [DEPTH][128]
    const float* __restrict__ fc3_w,      // [DEPTH][2][128]
    const float* __restrict__ fc3_b,      // [DEPTH][2]
    const float* __restrict__ lsf,        // [DEPTH][2]
    float* __restrict__ out)              // z:[N][2] then logdet:[N]
{
    // fc2 weights (f16, pre-scaled by 1/an2_scale[n]) in B-fragment order:
    // group g = ((nt*4+kf)*2 + p)*32 + lane  holds 8 halfs (K = kf*32 + (lane>=16?8:0) + p*16 + j)
    __shared__ _Float16 wbuf[32 * 2 * 32 * 8];          // 32768 B
    __shared__ float    alpha1_s[WIDTH];                // fc1 slope  (per feature)
    __shared__ float    beta1_s[WIDTH];                 // fc1 offset (per feature)
    __shared__ float    b2_s[WIDTH];                    // (fc2_b - an2_bias)/an2_scale
    __shared__ _Float16 g_s[2 * WIDTH];                 // fc3 rows * exp(lsf), f16
    __shared__ _Float16 abuf[WAVES_PER_BLOCK][16 * ABUF_STRIDE]; // h2 D->A relayout tile
    __shared__ float    f3buf[WAVES_PER_BLOCK][16 * 2];          // t / s columns

    const int tid  = threadIdx.x;
    const int lane = tid & 31;
    const int wave = tid >> 5;
    const int nl   = lane & 15;     // sample-row (scalar path) / N column (WMMA path)
    const int hi   = lane >> 4;     // half-wave select

    _Float16* aw  = abuf[wave];
    float*    fp  = f3buf[wave];

    // Per-wave per-tile sample state. Lanes 16..31 mirror lanes 0..15 (same sample nl),
    // which is exactly the duplication the A-fragment layout wants.
    float    z0[TILES_PER_WAVE], z1[TILES_PER_WAVE], ldet[TILES_PER_WAVE];
    unsigned row[TILES_PER_WAVE];

    #pragma unroll
    for (int tt = 0; tt < TILES_PER_WAVE; ++tt) {
        unsigned tile = blockIdx.x * TILES_PER_BLOCK + wave * TILES_PER_WAVE + tt;
        unsigned r = tile * 16u + (unsigned)nl;
        row[tt] = r;
        float2 xv = ((const float2*)x)[r];
        z0[tt] = xv.x; z1[tt] = xv.y; ldet[tt] = 0.0f;
    }

    #pragma unroll 1
    for (int i = 0; i < DEPTH; ++i) {
        // ------------------- stage per-layer parameters & weights -------------------
        if (tid < WIDTH) {
            const int f = tid;
            const float inv = 1.0f / an1_scale[i * WIDTH + f];
            alpha1_s[f] = fc1_w[i * WIDTH + f] * inv;
            beta1_s[f]  = (fc1_b[i * WIDTH + f] - an1_bias[i * WIDTH + f]) * inv;
        } else {
            const int n = tid - WIDTH;
            const float inv = 1.0f / an2_scale[i * WIDTH + n];
            b2_s[n] = (fc2_b[i * WIDTH + n] - an2_bias[i * WIDTH + n]) * inv;
        }
        {   // fc3 weight rows folded with exp(log_scale_factor)
            const int c = tid >> 7, k = tid & 127;
            g_s[tid] = (_Float16)(fc3_w[(i * 2 + c) * WIDTH + k] * __expf(lsf[i * 2 + c]));
        }
        // fc2 f32 -> f16 B-fragments, columns pre-scaled by 1/an2_scale[n]
        #pragma unroll
        for (int it = 0; it < 8; ++it) {
            const int g    = tid + it * 256;          // 0..2047
            const int frag = g >> 6;                  // nt*4 + kf
            const int p    = (g >> 5) & 1;
            const int ln   = g & 31;
            const int nt = frag >> 2, kf = frag & 3;
            const int n  = nt * 16 + (ln & 15);
            const int kb = kf * 32 + ((ln >> 4) << 3) + p * 16;
            const float4* wp = (const float4*)(fc2_w + (size_t)(i * WIDTH + n) * WIDTH + kb);
            const float4 w0 = wp[0], w1 = wp[1];
            const float sc = 1.0f / an2_scale[i * WIDTH + n];
            union { _Float16 hh[8]; int4 q; } u;
            u.hh[0] = (_Float16)(w0.x * sc); u.hh[1] = (_Float16)(w0.y * sc);
            u.hh[2] = (_Float16)(w0.z * sc); u.hh[3] = (_Float16)(w0.w * sc);
            u.hh[4] = (_Float16)(w1.x * sc); u.hh[5] = (_Float16)(w1.y * sc);
            u.hh[6] = (_Float16)(w1.z * sc); u.hh[7] = (_Float16)(w1.w * sc);
            *(int4*)(wbuf + (size_t)g * 8) = u.q;
        }
        __syncthreads();

        // uniform per-layer scalars
        const float ab0 = an_bias[i * 2 + 0], ab1 = an_bias[i * 2 + 1];
        const float as0 = an_scale[i * 2 + 0], as1 = an_scale[i * 2 + 1];
        const float c00 = conv_w[i * 4 + 0], c01 = conv_w[i * 4 + 1];
        const float c10 = conv_w[i * 4 + 2], c11 = conv_w[i * 4 + 3];
        const float det = c00 * c11 - c01 * c10;
        const float ldc = -__logf(fabsf(as0)) - __logf(fabsf(as1)) + __logf(fabsf(det));
        const float bb0 = fc3_b[i * 2 + 0] * __expf(lsf[i * 2 + 0]);
        const float bb1 = fc3_b[i * 2 + 1] * __expf(lsf[i * 2 + 1]);
        const float cbias = (nl == 0) ? bb0 : ((nl == 1) ? bb1 : 0.0f);

        #pragma unroll
        for (int tt = 0; tt < TILES_PER_WAVE; ++tt) {
            // actnorm + invertible 1x1 conv (2x2) per sample
            const float zna = (z0[tt] - ab0) / as0;
            const float znb = (z1[tt] - ab1) / as1;
            const float za  = c00 * zna + c01 * znb;   // z_a
            const float zb  = c10 * zna + c11 * znb;   // z_b

            // ---- h1 = relu(zb*alpha + beta): built straight into A fragments ----
            FragAB A[4];
            #pragma unroll
            for (int kf = 0; kf < 4; ++kf) {
                const int cbase = kf * 32 + hi * 8;
                #pragma unroll
                for (int hlf = 0; hlf < 2; ++hlf) {
                    const int f0 = cbase + hlf * 16;
                    const float4 av0 = *(const float4*)(alpha1_s + f0);
                    const float4 av1 = *(const float4*)(alpha1_s + f0 + 4);
                    const float4 bv0 = *(const float4*)(beta1_s + f0);
                    const float4 bv1 = *(const float4*)(beta1_s + f0 + 4);
                    const float aa[8] = {av0.x, av0.y, av0.z, av0.w, av1.x, av1.y, av1.z, av1.w};
                    const float bb[8] = {bv0.x, bv0.y, bv0.z, bv0.w, bv1.x, bv1.y, bv1.z, bv1.w};
                    #pragma unroll
                    for (int j = 0; j < 8; ++j)
                        A[kf].h[hlf * 8 + j] = (_Float16)fmaxf(zb * aa[j] + bb[j], 0.0f);
                }
            }

            // ---- fc2 via WMMA; fc3 accumulated via WMMA as h2 tiles complete ----
            v8f d;
            #pragma unroll
            for (int r = 0; r < 8; ++r) d[r] = cbias;   // fc3 accumulator (+bias)

            #pragma unroll
            for (int np = 0; np < 4; ++np) {            // pairs of n-tiles
                #pragma unroll
                for (int half = 0; half < 2; ++half) {
                    const int nt = np * 2 + half;
                    const float b2n = b2_s[nt * 16 + nl];
                    v8f c;
                    #pragma unroll
                    for (int r = 0; r < 8; ++r) c[r] = b2n;
                    #pragma unroll
                    for (int kf = 0; kf < 4; ++kf) {
                        FragAB B;
                        const int fb = (nt * 4 + kf) * 64 + lane;
                        B.q[0] = *(const int4*)(wbuf + (size_t)fb * 8);
                        B.q[1] = *(const int4*)(wbuf + (size_t)(fb + 32) * 8);
                        c = __builtin_amdgcn_wmma_f32_16x16x32_f16(
                                false, A[kf].v, false, B.v, (short)0, c, false, false);
                    }
                    // h2 = relu(D); stash into the relayout tile (D-layout -> row major)
                    const int fl = half * 16 + nl;
                    #pragma unroll
                    for (int r = 0; r < 8; ++r)
                        aw[(r + hi * 8) * ABUF_STRIDE + fl] = (_Float16)fmaxf(c[r], 0.0f);
                }
                // both 16-feature halves of k-frag np are ready: fc3 partial product
                FragAB A2, G;
                A2.q[0] = *(const int4*)(aw + nl * ABUF_STRIDE + hi * 8);
                A2.q[1] = *(const int4*)(aw + nl * ABUF_STRIDE + hi * 8 + 16);
                {
                    const int cidx = (nl < 2) ? nl : 0;
                    const int gb = cidx * WIDTH + np * 32 + hi * 8;
                    G.q[0] = *(const int4*)(g_s + gb);
                    G.q[1] = *(const int4*)(g_s + gb + 16);
                    if (nl >= 2) { G.q[0] = make_int4(0, 0, 0, 0); G.q[1] = make_int4(0, 0, 0, 0); }
                }
                d = __builtin_amdgcn_wmma_f32_16x16x32_f16(
                        false, A2.v, false, G.v, (short)0, d, false, false);
            }

            // ---- extract t (col 0) and pre-sigmoid s (col 1) per sample ----
            if (nl < 2) {
                #pragma unroll
                for (int r = 0; r < 8; ++r) fp[(r + hi * 8) * 2 + nl] = d[r];
            }
            const float t  = fp[nl * 2 + 0];
            const float sp = fp[nl * 2 + 1];
            const float s  = 1.0f / (1.0f + __expf(-(sp + 2.0f)));
            z0[tt] = s * za + t;
            z1[tt] = zb;
            ldet[tt] += ldc + __logf(s);
        }
        __syncthreads();   // waves done with this layer's wbuf/params before restage
    }

    // ------------------------------- store results -------------------------------
    if (lane < 16) {
        #pragma unroll
        for (int tt = 0; tt < TILES_PER_WAVE; ++tt) {
            out[2u * row[tt] + 0u] = z0[tt];
            out[2u * row[tt] + 1u] = z1[tt];
            out[2u * NROWS + row[tt]] = ldet[tt];
        }
    }
}

extern "C" void kernel_launch(void* const* d_in, const int* in_sizes, int n_in,
                              void* d_out, int out_size, void* d_ws, size_t ws_size,
                              hipStream_t stream) {
    const float* x    = (const float*)d_in[0];
    const float* ans  = (const float*)d_in[1];
    const float* anb  = (const float*)d_in[2];
    const float* cw   = (const float*)d_in[3];
    const float* f1w  = (const float*)d_in[4];
    const float* f1b  = (const float*)d_in[5];
    const float* a1s  = (const float*)d_in[6];
    const float* a1b  = (const float*)d_in[7];
    const float* f2w  = (const float*)d_in[8];
    const float* f2b  = (const float*)d_in[9];
    const float* a2s  = (const float*)d_in[10];
    const float* a2b  = (const float*)d_in[11];
    const float* f3wp = (const float*)d_in[12];
    const float* f3b  = (const float*)d_in[13];
    const float* lsfp = (const float*)d_in[14];
    float* outp = (float*)d_out;

    const unsigned blocks = NROWS / (16u * TILES_PER_BLOCK);   // 4096
    flow_wmma_kernel<<<dim3(blocks), dim3(256), 0, stream>>>(
        x, ans, anb, cw, f1w, f1b, a1s, a1b, f2w, f2b, a2s, a2b, f3wp, f3b, lsfp, outp);
}
